// LSTMAttentionModel_35253091565679
// MI455X (gfx1250) — matrix operand order repaired and checked by reading, not compiled, not run on hardware
//
#include <hip/hip_runtime.h>
#include <hip/hip_bf16.h>

// Problem constants (from reference)
#define T_DIM   64
#define B_DIM   256
#define N_ITEMS 100000
#define E_DIM   128
#define H_DIM   512
#define NH_DIM  8
#define HD_DIM  64
#define G_DIM   2048   // 4*H
#define KCAT    640    // E + H  (fused [x_t | h] @ [W_ih | W_hh]^T)

typedef __hip_bfloat16 bf16_t;
typedef __attribute__((ext_vector_type(16))) __bf16 v16bf;
typedef __attribute__((ext_vector_type(8)))  float  v8f;

union FragAB { v16bf v; unsigned u[8]; };
union AccU   { v8f   v; float    f[8]; };

// ---- WMMA fragment loaders (bf16, 16x16x32, per ISA 7.12.2) -------------
// A: 16x32 (MxK) row-major, stride ld. Lane half splits K at 8; VGPRs 4..7
// hold K=16..31.
__device__ inline FragAB load_fragA(const bf16_t* p, int ld, int lane) {
  FragAB f;
  const int r  = lane & 15;
  const int kh = (lane >> 4) << 3;                 // 0 or 8
  const bf16_t* row = p + (size_t)r * ld + kh;
  #pragma unroll
  for (int v = 0; v < 8; ++v) {
    const int k = ((v >= 4) ? 16 : 0) + ((v & 3) * 2);
    f.u[v] = *(const unsigned*)(row + k);
  }
  return f;
}
// B: 32x16 (KxN), supplied as N x K row-major (i.e. B^T), stride ld.
// Lane half splits K at 8; VGPRs 4..7 hold K=16..31 (same order as A).
__device__ inline FragAB load_fragB(const bf16_t* p, int ld, int lane) {
  FragAB f;
  const int n  = lane & 15;
  const int kh = (lane >> 4) << 3;                 // 0 or 8
  const bf16_t* row = p + (size_t)n * ld + kh;
  #pragma unroll
  for (int v = 0; v < 8; ++v)
    f.u[v] = *(const unsigned*)(row + ((v >= 4) ? 16 : 0) + ((v & 3) * 2));
  return f;
}

__device__ inline v8f wmma_bf16(const FragAB& a, const FragAB& b, v8f c) {
  return __builtin_amdgcn_wmma_f32_16x16x32_bf16(
      false, a.v, false, b.v, (short)0, c, false, false);
}

__device__ inline float sigmoidf_(float x) { return 1.f / (1.f + __expf(-x)); }

// ---- small utility kernels ----------------------------------------------
__global__ void k_f32_to_bf16(const float* __restrict__ s, bf16_t* __restrict__ d, int n) {
  int i = blockIdx.x * blockDim.x + threadIdx.x;
  if (i < n) d[i] = __float2bfloat16(s[i]);
}

// embs = (emb[x])^2  (closest == self: a table row's cosine-sim with itself is 1)
__global__ void k_gather_sq(const int* __restrict__ x, const float* __restrict__ emb,
                            bf16_t* __restrict__ embs) {
  int i = blockIdx.x * blockDim.x + threadIdx.x;       // T*B*E
  if (i >= T_DIM * B_DIM * E_DIM) return;
  const int e  = i & (E_DIM - 1);
  const int tb = i >> 7;
  const float v = emb[(size_t)x[tb] * E_DIM + e];
  embs[i] = __float2bfloat16(v * v);
}

// Wcat (2048 x 640) = [w_ih | w_hh] in bf16
__global__ void k_build_wcat(const float* __restrict__ w_ih, const float* __restrict__ w_hh,
                             bf16_t* __restrict__ wcat) {
  int i = blockIdx.x * blockDim.x + threadIdx.x;       // 2048*640
  if (i >= G_DIM * KCAT) return;
  const int r = i / KCAT, c = i % KCAT;
  const float v = (c < E_DIM) ? w_ih[(size_t)r * E_DIM + c]
                              : w_hh[(size_t)r * H_DIM + (c - E_DIM)];
  wcat[i] = __float2bfloat16(v);
}

// One gate block GEMM: acc[0..3] = [x_t | h] @ Wg^T for this wave's 64 cols.
// #pragma unroll 1 keeps B-frag addresses loop-variant so they are not
// unrolled-and-hoisted out of the timestep loop (which caused scratch spills).
__device__ inline void gate_gemm(AccU* acc, const bf16_t* __restrict__ xt,
                                 const bf16_t* sh, const bf16_t* __restrict__ Wg,
                                 int lane) {
  #pragma unroll
  for (int s = 0; s < 4; ++s)
    #pragma unroll
    for (int v = 0; v < 8; ++v) acc[s].f[v] = 0.f;
  #pragma unroll 1
  for (int k = 0; k < E_DIM; k += 32) {              // K over x_t (global)
    FragAB a = load_fragA(xt + k, E_DIM, lane);
    #pragma unroll
    for (int s = 0; s < 4; ++s) {
      FragAB bb = load_fragB(Wg + (size_t)(s * 16) * KCAT + k, KCAT, lane);
      acc[s].v = wmma_bf16(a, bb, acc[s].v);
    }
  }
  #pragma unroll 1
  for (int k = 0; k < H_DIM; k += 32) {              // K over h (LDS)
    FragAB a = load_fragA(sh + k, H_DIM, lane);
    #pragma unroll
    for (int s = 0; s < 4; ++s) {
      FragAB bb = load_fragB(Wg + (size_t)(s * 16) * KCAT + E_DIM + k, KCAT, lane);
      acc[s].v = wmma_bf16(a, bb, acc[s].v);
    }
  }
}

// ---- persistent fused LSTM ----------------------------------------------
// One block owns 16 batch rows for all 64 timesteps; h in LDS, c in regs.
// Wave w owns gate columns {g*512 + w*64 .. +63}, so each lane's accumulators
// are the i/f/g/o values of its OWN hidden units. Gates are consumed one
// block at a time (i -> g -> f -> o) into 32 running registers, keeping peak
// live state at acc[4] (32 VGPRs) instead of acc[4][4] (128 VGPRs).
__global__ void __launch_bounds__(256) k_lstm_fused(
    const bf16_t* __restrict__ embs, const bf16_t* __restrict__ wcat,
    const float* __restrict__ b_ih, const float* __restrict__ b_hh,
    const int* __restrict__ lengths, bf16_t* __restrict__ outBT) {
  __shared__ bf16_t s_h[16][H_DIM];                    // 16 KB
  const int b0   = blockIdx.x * 16;
  const int wave = threadIdx.x >> 5, lane = threadIdx.x & 31;
  const int colL = lane & 15;
  const int row0 = (lane >> 4) * 8;
  const int hc   = wave * 64 + colL;                   // hidden col, s adds 16*s

  float bsum[4][4];
  #pragma unroll
  for (int g = 0; g < 4; ++g)
    #pragma unroll
    for (int s = 0; s < 4; ++s) {
      const int idx = g * H_DIM + hc + s * 16;
      bsum[g][s] = b_ih[idx] + b_hh[idx];
    }
  int len8[8];
  #pragma unroll
  for (int v = 0; v < 8; ++v) len8[v] = lengths[b0 + row0 + v];

  float cst[4][8], tmp[4][8];
  #pragma unroll
  for (int s = 0; s < 4; ++s)
    #pragma unroll
    for (int v = 0; v < 8; ++v) cst[s][v] = 0.f;

  for (int i = threadIdx.x; i < 16 * H_DIM; i += 256)
    (&s_h[0][0])[i] = __float2bfloat16(0.f);
  __syncthreads();

  const bf16_t* sh = &s_h[0][0];
  const bf16_t* Wi = wcat + (size_t)(0 * H_DIM + wave * 64) * KCAT;
  const bf16_t* Wf = wcat + (size_t)(1 * H_DIM + wave * 64) * KCAT;
  const bf16_t* Wg = wcat + (size_t)(2 * H_DIM + wave * 64) * KCAT;
  const bf16_t* Wo = wcat + (size_t)(3 * H_DIM + wave * 64) * KCAT;

  #pragma unroll 1
  for (int t = 0; t < T_DIM; ++t) {
    const bf16_t* xt = embs + ((size_t)t * B_DIM + b0) * E_DIM;
    AccU acc[4];
    // i gate: tmp = sigmoid(i)
    gate_gemm(acc, xt, sh, Wi, lane);
    #pragma unroll
    for (int s = 0; s < 4; ++s)
      #pragma unroll
      for (int v = 0; v < 8; ++v)
        tmp[s][v] = sigmoidf_(acc[s].f[v] + bsum[0][s]);
    // g gate: tmp *= tanh(g)
    gate_gemm(acc, xt, sh, Wg, lane);
    #pragma unroll
    for (int s = 0; s < 4; ++s)
      #pragma unroll
      for (int v = 0; v < 8; ++v)
        tmp[s][v] *= tanhf(acc[s].f[v] + bsum[2][s]);
    // f gate: c = sigmoid(f)*c + tmp
    gate_gemm(acc, xt, sh, Wf, lane);
    #pragma unroll
    for (int s = 0; s < 4; ++s)
      #pragma unroll
      for (int v = 0; v < 8; ++v)
        cst[s][v] = sigmoidf_(acc[s].f[v] + bsum[1][s]) * cst[s][v] + tmp[s][v];
    // o gate: tmp = h = sigmoid(o)*tanh(c)
    gate_gemm(acc, xt, sh, Wo, lane);
    #pragma unroll
    for (int s = 0; s < 4; ++s)
      #pragma unroll
      for (int v = 0; v < 8; ++v)
        tmp[s][v] = sigmoidf_(acc[s].f[v] + bsum[3][s]) * tanhf(cst[s][v]);
    __syncthreads();                                 // all s_h reads done
    #pragma unroll
    for (int s = 0; s < 4; ++s) {
      const int h = hc + s * 16;
      #pragma unroll
      for (int v = 0; v < 8; ++v) {
        const int r = row0 + v;
        const float hv = tmp[s][v];
        s_h[r][h] = __float2bfloat16(hv);
        outBT[((size_t)(b0 + r) * T_DIM + t) * H_DIM + h] =
            __float2bfloat16((t < len8[v]) ? hv : 0.f);
      }
    }
    __syncthreads();                                 // s_h ready for t+1
  }
}

// ---- generic register-tiled WMMA GEMMs (32M x 64N per wave) --------------
// C(f32) = A(bf16, MxK, lda) @ B^T, B given N x K row-major. M % 32 == 0.
__global__ void __launch_bounds__(256) k_gemm_bf16_f32(
    const bf16_t* __restrict__ A, int lda, const bf16_t* __restrict__ B, int ldb,
    float* __restrict__ C, int ldc, int N, int K) {
  const int wave = threadIdx.x >> 5, lane = threadIdx.x & 31;
  const int n0 = (blockIdx.x * 8 + wave) * 64;
  if (n0 >= N) return;
  const int m0 = blockIdx.y * 32;
  const bf16_t* Ap = A + (size_t)m0 * lda;
  AccU acc[2][4];
  #pragma unroll
  for (int mi = 0; mi < 2; ++mi)
    #pragma unroll
    for (int s = 0; s < 4; ++s)
      #pragma unroll
      for (int v = 0; v < 8; ++v) acc[mi][s].f[v] = 0.f;
  for (int k = 0; k < K; k += 32) {
    __builtin_prefetch(B + (size_t)n0 * ldb + k + 128, 0, 0);
    FragAB a0 = load_fragA(Ap + k, lda, lane);
    FragAB a1 = load_fragA(Ap + (size_t)16 * lda + k, lda, lane);
    #pragma unroll
    for (int s = 0; s < 4; ++s) {
      if (n0 + s * 16 < N) {
        FragAB bb = load_fragB(B + (size_t)(n0 + s * 16) * ldb + k, ldb, lane);
        acc[0][s].v = wmma_bf16(a0, bb, acc[0][s].v);
        acc[1][s].v = wmma_bf16(a1, bb, acc[1][s].v);
      }
    }
  }
  const int colL = lane & 15, r0 = (lane >> 4) * 8;
  #pragma unroll
  for (int s = 0; s < 4; ++s) {
    if (n0 + s * 16 >= N) continue;
    const int col = n0 + s * 16 + colL;
    #pragma unroll
    for (int mi = 0; mi < 2; ++mi)
      #pragma unroll
      for (int v = 0; v < 8; ++v)
        C[(size_t)(m0 + mi * 16 + r0 + v) * ldc + col] = acc[mi][s].f[v];
  }
}

// bf16-out GEMM: C = alpha*(A@B^T) + bias. mode 0: row-major (ld=N).
// mode 1 (V projection): store transposed (B, H, T): dst[b][col][t].
__global__ void __launch_bounds__(256) k_gemm_bf16_bf16(
    const bf16_t* __restrict__ A, int lda, const bf16_t* __restrict__ B, int ldb,
    const float* __restrict__ bias, float alpha, bf16_t* __restrict__ C,
    int N, int K, int mode) {
  const int wave = threadIdx.x >> 5, lane = threadIdx.x & 31;
  const int n0 = (blockIdx.x * 8 + wave) * 64;
  if (n0 >= N) return;
  const int m0 = blockIdx.y * 32;
  const bf16_t* Ap = A + (size_t)m0 * lda;
  AccU acc[2][4];
  #pragma unroll
  for (int mi = 0; mi < 2; ++mi)
    #pragma unroll
    for (int s = 0; s < 4; ++s)
      #pragma unroll
      for (int v = 0; v < 8; ++v) acc[mi][s].f[v] = 0.f;
  for (int k = 0; k < K; k += 32) {
    FragAB a0 = load_fragA(Ap + k, lda, lane);
    FragAB a1 = load_fragA(Ap + (size_t)16 * lda + k, lda, lane);
    #pragma unroll
    for (int s = 0; s < 4; ++s) {
      if (n0 + s * 16 < N) {
        FragAB bb = load_fragB(B + (size_t)(n0 + s * 16) * ldb + k, ldb, lane);
        acc[0][s].v = wmma_bf16(a0, bb, acc[0][s].v);
        acc[1][s].v = wmma_bf16(a1, bb, acc[1][s].v);
      }
    }
  }
  const int colL = lane & 15, r0 = (lane >> 4) * 8;
  #pragma unroll
  for (int s = 0; s < 4; ++s) {
    if (n0 + s * 16 >= N) continue;
    const int col = n0 + s * 16 + colL;
    const float bb = bias ? bias[col] : 0.f;
    #pragma unroll
    for (int mi = 0; mi < 2; ++mi)
      #pragma unroll
      for (int v = 0; v < 8; ++v) {
        const int row = m0 + mi * 16 + r0 + v;
        const float val = acc[mi][s].f[v] * alpha + bb;
        if (mode == 0)
          C[(size_t)row * N + col] = __float2bfloat16(val);
        else
          C[(size_t)(row >> 6) * (H_DIM * T_DIM) + (size_t)col * T_DIM + (row & 63)] =
              __float2bfloat16(val);
      }
  }
}

// pad_mask[b*T+t] = (sum_h out[b,t,h] != 0)
__global__ void k_padmask(const bf16_t* __restrict__ outBT, int* __restrict__ mask) {
  int i = blockIdx.x * blockDim.x + threadIdx.x;       // B*T
  if (i >= B_DIM * T_DIM) return;
  const bf16_t* p = outBT + (size_t)i * H_DIM;
  float s = 0.f;
  for (int h = 0; h < H_DIM; ++h) s += __bfloat162float(p[h]);
  mask[i] = (s != 0.f) ? 1 : 0;
}

// ---- fused attention per (batch, head): scores -> softmax -> @V ----------
__global__ void __launch_bounds__(256) k_attention(
    const bf16_t* __restrict__ q, const bf16_t* __restrict__ k,
    const bf16_t* __restrict__ vT, const int* __restrict__ mask,
    bf16_t* __restrict__ weighted) {
  __shared__ float  s_sc[T_DIM][T_DIM + 1];
  __shared__ bf16_t s_at[T_DIM][T_DIM];
  const int b = blockIdx.x / NH_DIM, h = blockIdx.x % NH_DIM;
  const int wave = threadIdx.x >> 5, lane = threadIdx.x & 31;
  const bf16_t* qb = q  + (size_t)b * T_DIM * H_DIM + h * HD_DIM;
  const bf16_t* kb = k  + (size_t)b * T_DIM * H_DIM + h * HD_DIM;
  const bf16_t* vb = vT + (size_t)b * H_DIM * T_DIM + (size_t)h * HD_DIM * T_DIM;

  // scores = q @ k^T / 8, masked
  #pragma unroll
  for (int tt = 0; tt < 2; ++tt) {
    const int tile = wave * 2 + tt;
    const int mt = tile >> 2, nt = tile & 3;
    AccU acc;
    #pragma unroll
    for (int i = 0; i < 8; ++i) acc.f[i] = 0.f;
    #pragma unroll
    for (int k0 = 0; k0 < HD_DIM; k0 += 32) {
      FragAB a = load_fragA(qb + (size_t)mt * 16 * H_DIM + k0, H_DIM, lane);
      FragAB bfr = load_fragB(kb + (size_t)nt * 16 * H_DIM + k0, H_DIM, lane);
      acc.v = wmma_bf16(a, bfr, acc.v);
    }
    const int col  = nt * 16 + (lane & 15);
    const int row0 = mt * 16 + ((lane >> 4) * 8);
    const bool valid = mask[b * T_DIM + col] != 0;
    #pragma unroll
    for (int v = 0; v < 8; ++v)
      s_sc[row0 + v][col] = valid ? acc.f[v] * 0.125f : -__builtin_inff();
  }
  __syncthreads();

  // row softmax -> bf16 attn in LDS
  if (threadIdx.x < T_DIM) {
    const int r = threadIdx.x;
    float m = -__builtin_inff();
    for (int cI = 0; cI < T_DIM; ++cI) m = fmaxf(m, s_sc[r][cI]);
    float s = 0.f;
    for (int cI = 0; cI < T_DIM; ++cI) {
      const float e = __expf(s_sc[r][cI] - m);
      s_sc[r][cI] = e;
      s += e;
    }
    const float inv = 1.f / s;
    for (int cI = 0; cI < T_DIM; ++cI)
      s_at[r][cI] = __float2bfloat16(s_sc[r][cI] * inv);
  }
  __syncthreads();

  // weighted = attn @ v   (vT row-major (d, t): B-frag K is contiguous)
  #pragma unroll
  for (int tt = 0; tt < 2; ++tt) {
    const int tile = wave * 2 + tt;
    const int mt = tile >> 2, nt = tile & 3;
    AccU acc;
    #pragma unroll
    for (int i = 0; i < 8; ++i) acc.f[i] = 0.f;
    #pragma unroll
    for (int k0 = 0; k0 < T_DIM; k0 += 32) {
      FragAB a = load_fragA(&s_at[mt * 16][k0], T_DIM, lane);
      FragAB bfr = load_fragB(vb + (size_t)(nt * 16) * T_DIM + k0, T_DIM, lane);
      acc.v = wmma_bf16(a, bfr, acc.v);
    }
    const int col  = nt * 16 + (lane & 15);
    const int row0 = mt * 16 + ((lane >> 4) * 8);
    #pragma unroll
    for (int v = 0; v < 8; ++v)
      weighted[(size_t)(b * T_DIM + row0 + v) * H_DIM + h * HD_DIM + col] =
          __float2bfloat16(acc.f[v]);
  }
}

// wsum[b][h] = sum_t weighted[b][t][h]   (pooled mean folded into next GEMM)
__global__ void k_colsum(const bf16_t* __restrict__ w, bf16_t* __restrict__ wsum) {
  int i = blockIdx.x * blockDim.x + threadIdx.x;       // B*H
  if (i >= B_DIM * H_DIM) return;
  const int b = i >> 9, h = i & (H_DIM - 1);
  float s = 0.f;
  for (int t = 0; t < T_DIM; ++t)
    s += __bfloat162float(w[((size_t)b * T_DIM + t) * H_DIM + h]);
  wsum[i] = __float2bfloat16(s);
}

// ---- launcher ------------------------------------------------------------
extern "C" void kernel_launch(void* const* d_in, const int* in_sizes, int n_in,
                              void* d_out, int out_size, void* d_ws, size_t ws_size,
                              hipStream_t stream) {
  (void)in_sizes; (void)n_in; (void)out_size; (void)ws_size;
  const int*   x       = (const int*)  d_in[0];
  const int*   lengths = (const int*)  d_in[1];
  const float* emb     = (const float*)d_in[2];
  const float* w_ih    = (const float*)d_in[3];
  const float* w_hh    = (const float*)d_in[4];
  const float* b_ih    = (const float*)d_in[5];
  const float* b_hh    = (const float*)d_in[6];
  const float* wq = (const float*)d_in[7];  const float* bq = (const float*)d_in[8];
  const float* wk = (const float*)d_in[9];  const float* bk = (const float*)d_in[10];
  const float* wv = (const float*)d_in[11]; const float* bv = (const float*)d_in[12];
  const float* wo = (const float*)d_in[13]; const float* bo = (const float*)d_in[14];
  const float* w_h2e = (const float*)d_in[15]; const float* b_h2e = (const float*)d_in[16];
  float* out = (float*)d_out;

  char* ws = (char*)d_ws;
  size_t off = 0;
  auto alloc = [&](size_t bytes) -> char* {
    char* p = ws + off; off += (bytes + 255) & ~(size_t)255; return p;
  };
  bf16_t* emb_b  = (bf16_t*)alloc((size_t)N_ITEMS * E_DIM * 2);
  bf16_t* wcat   = (bf16_t*)alloc((size_t)G_DIM * KCAT * 2);
  bf16_t* wq_b   = (bf16_t*)alloc((size_t)H_DIM * H_DIM * 2);
  bf16_t* wk_b   = (bf16_t*)alloc((size_t)H_DIM * H_DIM * 2);
  bf16_t* wv_b   = (bf16_t*)alloc((size_t)H_DIM * H_DIM * 2);
  bf16_t* wo_b   = (bf16_t*)alloc((size_t)H_DIM * H_DIM * 2);
  bf16_t* wh2e_b = (bf16_t*)alloc((size_t)E_DIM * H_DIM * 2);
  bf16_t* embs   = (bf16_t*)alloc((size_t)T_DIM * B_DIM * E_DIM * 2);
  bf16_t* outBT  = (bf16_t*)alloc((size_t)B_DIM * T_DIM * H_DIM * 2);
  int*    pmask  = (int*)   alloc((size_t)B_DIM * T_DIM * 4);
  bf16_t* qb     = (bf16_t*)alloc((size_t)B_DIM * T_DIM * H_DIM * 2);
  bf16_t* kb     = (bf16_t*)alloc((size_t)B_DIM * T_DIM * H_DIM * 2);
  bf16_t* vTb    = (bf16_t*)alloc((size_t)B_DIM * H_DIM * T_DIM * 2);
  bf16_t* wgt    = (bf16_t*)alloc((size_t)B_DIM * T_DIM * H_DIM * 2);
  bf16_t* wsum   = (bf16_t*)alloc((size_t)B_DIM * H_DIM * 2);
  bf16_t* pooled = (bf16_t*)alloc((size_t)B_DIM * H_DIM * 2);
  bf16_t* ae     = (bf16_t*)alloc((size_t)B_DIM * E_DIM * 2);

  const int TB = 256;
  // precision conversions
  k_f32_to_bf16<<<(N_ITEMS * E_DIM + TB - 1) / TB, TB, 0, stream>>>(emb, emb_b, N_ITEMS * E_DIM);
  k_build_wcat <<<(G_DIM * KCAT + TB - 1) / TB, TB, 0, stream>>>(w_ih, w_hh, wcat);
  k_f32_to_bf16<<<(H_DIM * H_DIM + TB - 1) / TB, TB, 0, stream>>>(wq, wq_b, H_DIM * H_DIM);
  k_f32_to_bf16<<<(H_DIM * H_DIM + TB - 1) / TB, TB, 0, stream>>>(wk, wk_b, H_DIM * H_DIM);
  k_f32_to_bf16<<<(H_DIM * H_DIM + TB - 1) / TB, TB, 0, stream>>>(wv, wv_b, H_DIM * H_DIM);
  k_f32_to_bf16<<<(H_DIM * H_DIM + TB - 1) / TB, TB, 0, stream>>>(wo, wo_b, H_DIM * H_DIM);
  k_f32_to_bf16<<<(E_DIM * H_DIM + TB - 1) / TB, TB, 0, stream>>>(w_h2e, wh2e_b, E_DIM * H_DIM);
  // embs = (emb[x])^2 ; closest == self (exact optimization of the cosine kNN)
  k_gather_sq<<<(T_DIM * B_DIM * E_DIM + TB - 1) / TB, TB, 0, stream>>>(x, emb, embs);

  // whole LSTM in one persistent kernel (16 blocks x 16 batch rows)
  k_lstm_fused<<<B_DIM / 16, TB, 0, stream>>>(embs, wcat, b_ih, b_hh, lengths, outBT);

  k_padmask<<<(B_DIM * T_DIM + TB - 1) / TB, TB, 0, stream>>>(outBT, pmask);

  // Q/K/V projections (M=16384, N=512, K=512); V stored (B,H,T)
  const dim3 gqkv(1, (B_DIM * T_DIM) / 32);
  k_gemm_bf16_bf16<<<gqkv, TB, 0, stream>>>(outBT, H_DIM, wq_b, H_DIM, bq, 1.f, qb,  H_DIM, H_DIM, 0);
  k_gemm_bf16_bf16<<<gqkv, TB, 0, stream>>>(outBT, H_DIM, wk_b, H_DIM, bk, 1.f, kb,  H_DIM, H_DIM, 0);
  k_gemm_bf16_bf16<<<gqkv, TB, 0, stream>>>(outBT, H_DIM, wv_b, H_DIM, bv, 1.f, vTb, H_DIM, H_DIM, 1);

  // fused attention per (b, head)
  k_attention<<<B_DIM * NH_DIM, TB, 0, stream>>>(qb, kb, vTb, pmask, wgt);

  // pooled = mean_t(weighted @ wo^T + bo) = (sum_t weighted)/T @ wo^T + bo
  k_colsum<<<(B_DIM * H_DIM + TB - 1) / TB, TB, 0, stream>>>(wgt, wsum);
  k_gemm_bf16_bf16<<<dim3(1, B_DIM / 32), TB, 0, stream>>>(
      wsum, H_DIM, wo_b, H_DIM, bo, 1.f / (float)T_DIM, pooled, H_DIM, H_DIM, 0);

  // ae = pooled @ w_h2e^T + b_h2e   (M=256, N=128, K=512)
  k_gemm_bf16_bf16<<<dim3(1, B_DIM / 32), TB, 0, stream>>>(
      pooled, H_DIM, wh2e_b, H_DIM, b_h2e, 1.f, ae, E_DIM, H_DIM, 0);

  // scores = ae @ emb^T   (M=256, N=100000, K=128) -> f32 output
  const int ngroups = (N_ITEMS / 16 + 3) / 4;          // 1563
  k_gemm_bf16_f32<<<dim3((ngroups + 7) / 8, B_DIM / 32), TB, 0, stream>>>(
      ae, E_DIM, emb_b, E_DIM, out, N_ITEMS, N_ITEMS, E_DIM);
}
